// CustomGNN1_64785286692985
// MI455X (gfx1250) — matrix operand order repaired and checked by reading, not compiled, not run on hardware
//
#include <hip/hip_runtime.h>
#include <hip/hip_bf16.h>

typedef __attribute__((ext_vector_type(16))) _Float16 v16h;
typedef __attribute__((ext_vector_type(8)))  float    v8f;

#define HDIM 32
#define DNDIM 16
#define DEDIM 8
#define KIN   40   // H + DE

// ---------------------------------------------------------------------------
// Monotonic float <-> uint mapping: order(float) == order(uint).
// Lets scatter-max use a single unconditional global_atomic_max_u32.
// flip(-inf) = 0x007FFFFF  (used as the "empty segment" init value).
// ---------------------------------------------------------------------------
#define FLIP_NEG_INF 0x007FFFFFu

__device__ __forceinline__ unsigned flip_f32(float f) {
    unsigned b = __float_as_uint(f);
    return b ^ (unsigned)(((int)b >> 31) | (int)0x80000000u);
}
__device__ __forceinline__ float unflip_f32(unsigned u) {
    unsigned m = (unsigned)(~((int)u >> 31)) | 0x80000000u;
    return __uint_as_float(u ^ m);
}

// ---------------------------------------------------------------------------
// h = x @ W_node + b_node   (N x 16) @ (16 x 32)  -- memory bound, VALU
// ---------------------------------------------------------------------------
__global__ __launch_bounds__(256)
void gnn_node_init(const float* __restrict__ x, const float* __restrict__ W,
                   const float* __restrict__ b, float* __restrict__ h, int N_) {
    int i = blockIdx.x * blockDim.x + threadIdx.x;
    if (i >= N_ * HDIM) return;
    int node = i >> 5;
    int n    = i & 31;
    const float* xr = x + (long)node * DNDIM;
    float acc = b[n];
#pragma unroll
    for (int k = 0; k < DNDIM; ++k) acc = fmaf(xr[k], W[k * HDIM + n], acc);
    h[i] = acc;
}

__global__ __launch_bounds__(256)
void gnn_fill(unsigned* __restrict__ p, int n, unsigned v) {
    int i = blockIdx.x * blockDim.x + threadIdx.x;
    if (i < n) p[i] = v;
}

// ---------------------------------------------------------------------------
// Message + scatter-max.  One wave handles a tile of 16 edges:
//   msg[16x32] = [h[src] | edge_attr][16x40] @ Wm[40x32] + bm
// K padded 40 -> 64 (two K=32 chunks), two N-tiles of 16.
// 4x v_wmma_f32_16x16x32_f16 per tile; scatter = 1 atomic_max_u32 per element
// (flipped-uint encoding, 32-bit element offsets), dst distributed by
// ds_bpermute (__shfl).
// ---------------------------------------------------------------------------
__global__ __launch_bounds__(256)
void gnn_message(const float* __restrict__ h,
                 const float* __restrict__ edge_attr,
                 const int* __restrict__ srcI,
                 const int* __restrict__ dstI,
                 const float* __restrict__ Wm,   // 40 x 32 (row-major K x N)
                 const float* __restrict__ bm,   // 32
                 unsigned* __restrict__ aggr,    // N x 32, pre-filled FLIP_NEG_INF
                 int E_) {
    const int lane    = threadIdx.x & 31;
    const int wave    = threadIdx.x >> 5;
    const int halfsel = lane >> 4;       // 0: lanes 0-15, 1: lanes 16-31
    const int l15     = lane & 15;
    const int kb      = 8 * halfsel;     // A-fragment K base per ISA layout

    // ---- B fragments: Wm -> f16, [K-chunk][N-tile], loaded once per wave.
    // B layout (16-bit, 32x16): lane holds col n = l15+16t; elem e -> K = 8*half + e (+8 if e>=8)
    v16h Bf[2][2];
#pragma unroll
    for (int c = 0; c < 2; ++c) {
#pragma unroll
        for (int t = 0; t < 2; ++t) {
            int n = l15 + 16 * t;
#pragma unroll
            for (int e = 0; e < 16; ++e) {
                int k = 32 * c + kb + e + (e >= 8 ? 8 : 0);
                float w = (k < KIN) ? Wm[k * HDIM + n] : 0.0f;
                Bf[c][t][e] = (_Float16)w;
            }
        }
    }
    float bias[2] = { bm[l15], bm[l15 + 16] };

    const int tiles  = (E_ + 15) >> 4;
    const int stride = gridDim.x * 8;
    for (int tile = blockIdx.x * 8 + wave; tile < tiles; tile += stride) {
        // prefetch next tile's streamed data (edge_index rows + edge_attr)
        long enext = (long)(tile + stride) * 16 + l15;
        if (enext < E_) {
            __builtin_prefetch(srcI + enext, 0, 0);
            __builtin_prefetch(dstI + enext, 0, 0);
            __builtin_prefetch(edge_attr + enext * DEDIM, 0, 0);
        }

        // ---- A fragments. Lane holds row m = l15 (both half-waves).
        int  e    = tile * 16 + l15;
        bool full = (tile * 16 + 16 <= E_);       // uniform per wave
        int  ec   = (e < E_) ? e : (E_ - 1);
        int  s    = srcI[ec];
        int  dstv = dstI[ec];                     // this lane's row -> dst node
        const float4* h4 = (const float4*)(h + (long)s * HDIM);

        // chunk0: K=0..31 from h[src] row.
        // elems 0..7 -> K = kb..kb+7 ; elems 8..15 -> K = kb+16..kb+23
        float4 r0 = h4[(kb >> 2) + 0];
        float4 r1 = h4[(kb >> 2) + 1];
        float4 r2 = h4[(kb >> 2) + 4];
        float4 r3 = h4[(kb >> 2) + 5];
        v16h A0, A1;
        A0[0]=(_Float16)r0.x; A0[1]=(_Float16)r0.y; A0[2]=(_Float16)r0.z; A0[3]=(_Float16)r0.w;
        A0[4]=(_Float16)r1.x; A0[5]=(_Float16)r1.y; A0[6]=(_Float16)r1.z; A0[7]=(_Float16)r1.w;
        A0[8]=(_Float16)r2.x; A0[9]=(_Float16)r2.y; A0[10]=(_Float16)r2.z; A0[11]=(_Float16)r2.w;
        A0[12]=(_Float16)r3.x; A0[13]=(_Float16)r3.y; A0[14]=(_Float16)r3.z; A0[15]=(_Float16)r3.w;

        // chunk1: global K = 32 + (kb + ...). Only K=32..39 (edge_attr) nonzero,
        // i.e. lower half-wave (kb==0), elems 0..7.
#pragma unroll
        for (int q = 0; q < 16; ++q) A1[q] = (_Float16)0.0f;
        if (halfsel == 0) {
            const float4* a4 = (const float4*)(edge_attr + (long)ec * DEDIM);
            float4 a0 = a4[0];
            float4 a1 = a4[1];
            A1[0]=(_Float16)a0.x; A1[1]=(_Float16)a0.y; A1[2]=(_Float16)a0.z; A1[3]=(_Float16)a0.w;
            A1[4]=(_Float16)a1.x; A1[5]=(_Float16)a1.y; A1[6]=(_Float16)a1.z; A1[7]=(_Float16)a1.w;
        }

#pragma unroll
        for (int t = 0; t < 2; ++t) {
            v8f C;
#pragma unroll
            for (int r = 0; r < 8; ++r) C[r] = bias[t];
            C = __builtin_amdgcn_wmma_f32_16x16x32_f16(false, A0, false, Bf[0][t],
                                                       (short)0, C, false, false);
            C = __builtin_amdgcn_wmma_f32_16x16x32_f16(false, A1, false, Bf[1][t],
                                                       (short)0, C, false, false);
            // D layout: lane col n = l15+16t, rows m = r + 8*halfsel
            int n = l15 + 16 * t;
            if (full) {                            // uniform fast path
#pragma unroll
                for (int r = 0; r < 8; ++r) {
                    int mr = r + 8 * halfsel;
                    int d  = __shfl(dstv, mr, 32); // ds_bpermute row -> all lanes
                    unsigned off = (unsigned)d * HDIM + (unsigned)n;  // 32-bit offs
                    atomicMax(aggr + off, flip_f32(C[r]));
                }
            } else {                               // guarded tail tile
#pragma unroll
                for (int r = 0; r < 8; ++r) {
                    int mr = r + 8 * halfsel;
                    int d  = __shfl(dstv, mr, 32);
                    unsigned off = (unsigned)d * HDIM + (unsigned)n;
                    if (tile * 16 + mr < E_)
                        atomicMax(aggr + off, flip_f32(C[r]));
                }
            }
        }
    }
}

// ---------------------------------------------------------------------------
// Update: h += relu( fix(unflip(aggr)) @ Wu + bu ), empty segment -> 0.
// One wave per 16-node tile; K=32 exactly one chunk; 2 N-tiles -> 2 WMMAs.
// Also re-initializes aggr to FLIP_NEG_INF for the next layer while the
// lines are hot in L2 (each element is read by exactly one lane, and loads
// issue before the re-init stores in program order -> race-free).
// ---------------------------------------------------------------------------
__global__ __launch_bounds__(256)
void gnn_update(float* __restrict__ h,
                unsigned* __restrict__ aggr,
                const float* __restrict__ Wu,   // 32 x 32
                const float* __restrict__ bu,   // 32
                int N_) {
    const int lane    = threadIdx.x & 31;
    const int wave    = threadIdx.x >> 5;
    const int halfsel = lane >> 4;
    const int l15     = lane & 15;
    const int kb      = 8 * halfsel;

    v16h Bf[2];
#pragma unroll
    for (int t = 0; t < 2; ++t) {
        int n = l15 + 16 * t;
#pragma unroll
        for (int e = 0; e < 16; ++e) {
            int k = kb + e + (e >= 8 ? 8 : 0);
            Bf[t][e] = (_Float16)Wu[k * HDIM + n];
        }
    }
    float bias[2] = { bu[l15], bu[l15 + 16] };

    const uint4 FILL = { FLIP_NEG_INF, FLIP_NEG_INF, FLIP_NEG_INF, FLIP_NEG_INF };

    const int tiles = (N_ + 15) >> 4;
    for (int tile = blockIdx.x * 8 + wave; tile < tiles; tile += gridDim.x * 8) {
        int node  = tile * 16 + l15;
        int nodec = (node < N_) ? node : (N_ - 1);
        uint4* g4 = (uint4*)(aggr + (unsigned)nodec * HDIM);
        uint4 r0 = g4[(kb >> 2) + 0];
        uint4 r1 = g4[(kb >> 2) + 1];
        uint4 r2 = g4[(kb >> 2) + 4];
        uint4 r3 = g4[(kb >> 2) + 5];
        // fused re-init of aggr for the next layer (free: lines hot in L2)
        g4[(kb >> 2) + 0] = FILL;
        g4[(kb >> 2) + 1] = FILL;
        g4[(kb >> 2) + 4] = FILL;
        g4[(kb >> 2) + 5] = FILL;
        unsigned av[16] = { r0.x, r0.y, r0.z, r0.w, r1.x, r1.y, r1.z, r1.w,
                            r2.x, r2.y, r2.z, r2.w, r3.x, r3.y, r3.z, r3.w };
        v16h A;
#pragma unroll
        for (int q = 0; q < 16; ++q) {
            unsigned u = av[q];
            float v = (u == FLIP_NEG_INF) ? 0.0f : unflip_f32(u);  // empty -> 0
            A[q] = (_Float16)v;
        }

#pragma unroll
        for (int t = 0; t < 2; ++t) {
            v8f C;
#pragma unroll
            for (int r = 0; r < 8; ++r) C[r] = bias[t];
            C = __builtin_amdgcn_wmma_f32_16x16x32_f16(false, A, false, Bf[t],
                                                       (short)0, C, false, false);
            int n = l15 + 16 * t;
#pragma unroll
            for (int r = 0; r < 8; ++r) {
                int mr = r + 8 * halfsel;
                int nd = tile * 16 + mr;
                if (nd < N_) {
                    float u = C[r];
                    u = (u > 0.0f) ? u : 0.0f;                 // ReLU
                    unsigned idx = (unsigned)nd * HDIM + (unsigned)n;
                    h[idx] = h[idx] + u;                       // residual bypass
                }
            }
        }
    }
}

// ---------------------------------------------------------------------------
// out = h @ W_post + b_post   (N x 32) @ (32 x 1)
// ---------------------------------------------------------------------------
__global__ __launch_bounds__(256)
void gnn_post(const float* __restrict__ h, const float* __restrict__ Wp,
              const float* __restrict__ bp, float* __restrict__ out, int N_) {
    int i = blockIdx.x * blockDim.x + threadIdx.x;
    if (i >= N_) return;
    const float* hr = h + (unsigned)i * HDIM;
    float acc = bp[0];
#pragma unroll
    for (int k = 0; k < HDIM; ++k) acc = fmaf(hr[k], Wp[k], acc);
    out[i] = acc;
}

// ---------------------------------------------------------------------------
extern "C" void kernel_launch(void* const* d_in, const int* in_sizes, int n_in,
                              void* d_out, int out_size, void* d_ws, size_t ws_size,
                              hipStream_t stream) {
    const float* x         = (const float*)d_in[0];
    const float* edge_attr = (const float*)d_in[1];
    const int*   edge_idx  = (const int*)  d_in[2];
    const float* W_node    = (const float*)d_in[3];
    const float* b_node    = (const float*)d_in[4];
    const float* Wm        = (const float*)d_in[5];
    const float* bm        = (const float*)d_in[6];
    const float* Wu        = (const float*)d_in[7];
    const float* bu        = (const float*)d_in[8];
    const float* W_post    = (const float*)d_in[9];
    const float* b_post    = (const float*)d_in[10];

    const int N_ = in_sizes[0] / DNDIM;      // 100000
    const int E_ = in_sizes[1] / DEDIM;      // 1600000
    const int L_ = in_sizes[6] / HDIM;       // 7

    const int* srcI = edge_idx;              // edge_index[0]
    const int* dstI = edge_idx + E_;         // edge_index[1]

    float*    hbuf = (float*)d_ws;                          // N x 32
    unsigned* aggr = (unsigned*)(hbuf + (size_t)N_ * HDIM); // N x 32 (flipped)

    gnn_node_init<<<(N_ * HDIM + 255) / 256, 256, 0, stream>>>(x, W_node, b_node, hbuf, N_);

    const int tilesE = (E_ + 15) / 16;
    const int tilesN = (N_ + 15) / 16;
    int gridE = (tilesE + 7) / 8; if (gridE > 2048) gridE = 2048;
    int gridN = (tilesN + 7) / 8; if (gridN > 2048) gridN = 2048;

    // single init of aggr; gnn_update re-initializes it each layer for free
    gnn_fill<<<(N_ * HDIM + 255) / 256, 256, 0, stream>>>(aggr, N_ * HDIM, FLIP_NEG_INF);

    for (int l = 0; l < L_; ++l) {
        gnn_message<<<gridE, 256, 0, stream>>>(hbuf, edge_attr, srcI, dstI,
                                               Wm + (size_t)l * KIN * HDIM,
                                               bm + (size_t)l * HDIM, aggr, E_);
        gnn_update<<<gridN, 256, 0, stream>>>(hbuf, aggr,
                                              Wu + (size_t)l * HDIM * HDIM,
                                              bu + (size_t)l * HDIM, N_);
    }

    gnn_post<<<(N_ + 255) / 256, 256, 0, stream>>>(hbuf, W_post, b_post, (float*)d_out, N_);
}
